// SparseAttentionModel_71030169141948
// MI455X (gfx1250) — matrix operand order repaired and checked by reading, not compiled, not run on hardware
//
#include <hip/hip_runtime.h>
#include <hip/hip_fp16.h>
#include <hip/hip_bf16.h>

typedef __attribute__((ext_vector_type(16))) _Float16 v16h;
typedef __attribute__((ext_vector_type(8)))  _Float16 v8h;
typedef __attribute__((ext_vector_type(8)))  float    v8f;

#define BBATCH 32
#define NSEQ   3072
#define DIMM   64
#define DH     48
#define NQKV   144      // 3*DH
#define CBS    12
#define CBD    576      // CBS*DH, == 18*32 (exact K tiling)
#define NBLK   256      // NSEQ / CBS
#define NCPAD  288      // 1 (mem) + 256 compressed rows, padded to 9*32
#define KPAD   64       // head dim padded for K=32 WMMA steps
#define WIN    6
#define SCALEF 0.14433756729740643f   // 48^-0.5
#define NEGI  (-1e30f)

union V16U { v16h v; struct { v8h lo, hi; } p; _Float16 h[16]; };

#define WMMA_F16(A, B, C) \
  __builtin_amdgcn_wmma_f32_16x16x32_f16(false, (A), false, (B), (short)0, (C), false, false)

// ---------------------------------------------------------------------------
// Kernel W: transpose + f16-convert weights into B-fragment layouts
//   WqkvT [144][64], WkcT [48][576], WvcT [48][576]
// ---------------------------------------------------------------------------
__global__ __launch_bounds__(256) void k_prepw(
    const float* __restrict__ Wqkv, const float* __restrict__ Wkc,
    const float* __restrict__ Wvc, _Float16* __restrict__ WqkvT,
    _Float16* __restrict__ WkcT, _Float16* __restrict__ WvcT)
{
  int e = blockIdx.x * 256 + threadIdx.x;
  if (e < NQKV * DIMM) {
    int n = e / DIMM, k = e % DIMM;
    WqkvT[e] = (_Float16)Wqkv[k * NQKV + n];
  } else if (e < NQKV * DIMM + DH * CBD) {
    int e2 = e - NQKV * DIMM;
    int d = e2 / CBD, kk = e2 % CBD;
    WkcT[e2] = (_Float16)Wkc[kk * DH + d];
  } else if (e < NQKV * DIMM + 2 * DH * CBD) {
    int e2 = e - NQKV * DIMM - DH * CBD;
    int d = e2 / CBD, kk = e2 % CBD;
    WvcT[e2] = (_Float16)Wvc[kk * DH + d];
  }
}

// ---------------------------------------------------------------------------
// Kernel A: embedding + RMSNorm (-> f16 xn) + gates
// ---------------------------------------------------------------------------
__global__ __launch_bounds__(256) void k_embed(
    const float* __restrict__ X, const float* __restrict__ w_emb,
    const float* __restrict__ b_emb, const float* __restrict__ gamma,
    _Float16* __restrict__ xnh, const float* __restrict__ Wg,
    const float* __restrict__ bg, float* __restrict__ gg)
{
  int t = blockIdx.x * 256 + threadIdx.x;        // t = b*NSEQ + i
  float x = X[t];
  float xn[DIMM];
  float ms = 0.f;
  #pragma unroll
  for (int d = 0; d < DIMM; ++d) {
    float v = x * w_emb[d] + b_emb[d];
    xn[d] = v; ms += v * v;
  }
  ms *= (1.0f / DIMM);
  float rn = rsqrtf(ms + 1e-12f);
  #pragma unroll
  for (int d = 0; d < DIMM; ++d) xn[d] *= rn * gamma[d];

  size_t xb = (size_t)t * DIMM;
  #pragma unroll
  for (int d = 0; d < DIMM; ++d) xnh[xb + d] = (_Float16)xn[d];

  #pragma unroll
  for (int c = 0; c < 3; ++c) {
    float a = bg[c];
    for (int d = 0; d < DIMM; ++d) a += xn[d] * Wg[d * 3 + c];
    gg[(size_t)t * 3 + c] = 1.0f / (1.0f + __expf(-a));
  }
}

// ---------------------------------------------------------------------------
// Kernel Q: qkv = xn @ Wqkv via WMMA (M=16 tokens, K=64, N=144), fused RoPE
//   (even/odd RoPE pairs are adjacent lanes in the D layout -> shfl_xor(1)),
//   and direct scatter of (k+k_pos)/(v+v_pos) into f16 block-flat layout.
// ---------------------------------------------------------------------------
__global__ __launch_bounds__(32) void k_qkv(
    const _Float16* __restrict__ xnh, const _Float16* __restrict__ WqkvT,
    const float* __restrict__ k_pos, const float* __restrict__ v_pos,
    _Float16* __restrict__ qh, float* __restrict__ qr, float* __restrict__ kr,
    float* __restrict__ vf, _Float16* __restrict__ kbh, _Float16* __restrict__ vbh)
{
  const int QT = NSEQ / 16;
  int b    = blockIdx.x / QT;
  int qt   = blockIdx.x % QT;
  int lane = threadIdx.x;
  int m  = lane & 15;
  int hi = lane >> 4;

  const _Float16* xb = xnh + ((size_t)b * NSEQ + qt * 16 + m) * DIMM;
  V16U a0, a1;
  a0.p.lo = *(const v8h*)(xb + hi * 16);
  a0.p.hi = *(const v8h*)(xb + 8 + hi * 16);
  a1.p.lo = *(const v8h*)(xb + 32 + hi * 16);
  a1.p.hi = *(const v8h*)(xb + 40 + hi * 16);

  v8f acc[9];
  #pragma unroll
  for (int nt = 0; nt < 9; ++nt) {
    const _Float16* wb = WqkvT + (size_t)(nt * 16 + m) * DIMM;
    v16h b0 = *(const v16h*)(wb + hi * 16);
    v16h b1 = *(const v16h*)(wb + 32 + hi * 16);
    v8f c = {};
    c = WMMA_F16(a0.v, b0, c);
    c = WMMA_F16(a1.v, b1, c);
    acc[nt] = c;
  }

  #pragma unroll
  for (int r = 0; r < 8; ++r) {
    int row = qt * 16 + r + 8 * hi;            // token index i
    size_t gi = (size_t)b * NSEQ + row;
    int jb = row / CBS, sblk = row % CBS;
    // ---- q columns 0..47: store raw f16 q + RoPE'd q ----
    #pragma unroll
    for (int nt = 0; nt < 3; ++nt) {
      int col = nt * 16 + m;
      float qv = acc[nt][r];
      qh[gi * KPAD + col] = (_Float16)qv;
      float part = __shfl_xor(qv, 1);
      int p = col >> 1;
      float inv = __powf(10000.0f, -(2.0f * p) / (float)DH);
      float sn, cs; __sincosf((float)row * inv, &sn, &cs);
      float o = ((m & 1) == 0) ? (qv * cs - part * sn) : (part * sn + qv * cs);
      qr[gi * DH + col] = o;
    }
    qh[gi * KPAD + 48 + m] = (_Float16)0.f;    // K pad 48..63
    // ---- k columns 48..95: RoPE'd k + (k + k_pos) into flat block f16 ----
    #pragma unroll
    for (int nt = 3; nt < 6; ++nt) {
      int col = nt * 16 + m - 48;
      float kv = acc[nt][r];
      float part = __shfl_xor(kv, 1);
      int p = col >> 1;
      float inv = __powf(10000.0f, -(2.0f * p) / (float)DH);
      float sn, cs; __sincosf((float)row * inv, &sn, &cs);
      float o = ((m & 1) == 0) ? (kv * cs - part * sn) : (part * sn + kv * cs);
      kr[gi * DH + col] = o;
      int e = sblk * DH + col;
      kbh[((size_t)b * NBLK + jb) * CBD + e] = (_Float16)(kv + k_pos[e]);
    }
    // ---- v columns 96..143: f32 v + (v + v_pos) into flat block f16 ----
    #pragma unroll
    for (int nt = 6; nt < 9; ++nt) {
      int col = nt * 16 + m - 96;
      float vv = acc[nt][r];
      vf[gi * DH + col] = vv;
      int e = sblk * DH + col;
      vbh[((size_t)b * NBLK + jb) * CBD + e] = (_Float16)(vv + v_pos[e]);
    }
  }
}

// ---------------------------------------------------------------------------
// Kernel B0: init mem rows + padding of compressed K/V tables
// ---------------------------------------------------------------------------
__global__ __launch_bounds__(NCPAD) void k_initc(
    const float* __restrict__ mem_k, const float* __restrict__ mem_v,
    _Float16* __restrict__ ckf, _Float16* __restrict__ cvfT)
{
  int b = blockIdx.x;
  int j = threadIdx.x;                       // 0..287
  _Float16* row = ckf + ((size_t)b * NCPAD + j) * KPAD;
  for (int d = DH; d < KPAD; ++d) row[d] = (_Float16)0.f;   // head-dim pad
  if (j == 0) {
    for (int d = 0; d < DH; ++d) row[d] = (_Float16)mem_k[d];
    for (int d = 0; d < DH; ++d) cvfT[((size_t)b * DH + d) * NCPAD] = (_Float16)mem_v[d];
  } else if (j > NBLK) {                     // j >= 257 : zero pad rows/cols
    for (int d = 0; d < DH; ++d) row[d] = (_Float16)0.f;
    for (int d = 0; d < DH; ++d) cvfT[((size_t)b * DH + d) * NCPAD + j] = (_Float16)0.f;
  }
}

// ---------------------------------------------------------------------------
// Kernel B: K/V block compression via WMMA
//   M = 16 compressed blocks/wave, K = 576 (18 steps), N = 48 (x2 for K and V)
// ---------------------------------------------------------------------------
__global__ __launch_bounds__(32) void k_compress(
    const _Float16* __restrict__ kbh, const _Float16* __restrict__ vbh,
    const _Float16* __restrict__ WkcT, const _Float16* __restrict__ WvcT,
    const float* __restrict__ bkc, const float* __restrict__ bvc,
    _Float16* __restrict__ ckf, _Float16* __restrict__ cvfT)
{
  const int JT = NBLK / 16;                  // 16 block-tiles per batch
  int b    = blockIdx.x / JT;
  int jt   = blockIdx.x % JT;
  int lane = threadIdx.x;
  int m  = lane & 15;
  int hi = lane >> 4;

  const _Float16* ka = kbh + ((size_t)b * NBLK + jt * 16 + m) * CBD;
  const _Float16* va = vbh + ((size_t)b * NBLK + jt * 16 + m) * CBD;

  v8f ak[3] = {v8f{}, v8f{}, v8f{}};
  v8f av[3] = {v8f{}, v8f{}, v8f{}};
  for (int t = 0; t < CBD / 32; ++t) {       // 18 K-steps
    V16U aK, aV;
    aK.p.lo = *(const v8h*)(ka + t * 32 + hi * 16);
    aK.p.hi = *(const v8h*)(ka + t * 32 + 8 + hi * 16);
    aV.p.lo = *(const v8h*)(va + t * 32 + hi * 16);
    aV.p.hi = *(const v8h*)(va + t * 32 + 8 + hi * 16);
    #pragma unroll
    for (int nt = 0; nt < 3; ++nt) {
      v16h bk = *(const v16h*)(WkcT + (size_t)(nt * 16 + m) * CBD + t * 32 + hi * 16);
      v16h bv = *(const v16h*)(WvcT + (size_t)(nt * 16 + m) * CBD + t * 32 + hi * 16);
      ak[nt] = WMMA_F16(aK.v, bk, ak[nt]);
      av[nt] = WMMA_F16(aV.v, bv, av[nt]);
    }
  }
  #pragma unroll
  for (int r = 0; r < 8; ++r) {
    int jb = jt * 16 + r + 8 * hi;
    #pragma unroll
    for (int nt = 0; nt < 3; ++nt) {
      int d = nt * 16 + m;
      ckf[((size_t)b * NCPAD + jb + 1) * KPAD + d] = (_Float16)(ak[nt][r] + bkc[d]);
      cvfT[((size_t)b * DH + d) * NCPAD + jb + 1]  = (_Float16)(av[nt][r] + bvc[d]);
    }
  }
}

// ---------------------------------------------------------------------------
// Kernel C: compressed attention (WMMA), softmax, top-block selection,
//           and c_out = P @ V (WMMA).   One wave per 16-query tile.
// ---------------------------------------------------------------------------
__global__ __launch_bounds__(32) void k_cattn(
    const _Float16* __restrict__ qh, const _Float16* __restrict__ ckf,
    const _Float16* __restrict__ cvfT, float* __restrict__ c_out,
    int* __restrict__ selJb, int* __restrict__ selOk)
{
  const int QT = NSEQ / 16;                  // 192 tiles per batch
  int b    = blockIdx.x / QT;
  int qt   = blockIdx.x % QT;
  int lane = threadIdx.x;
  int m  = lane & 15;
  int hi = lane >> 4;

  __shared__ float P[16][NCPAD];
  __shared__ float denomS[16];

  // ---- A fragments (16 query rows, K = 0..63 in two K=32 steps) ----
  const _Float16* qbase = qh + ((size_t)b * NSEQ + qt * 16 + m) * KPAD;
  V16U a0, a1;
  a0.p.lo = *(const v8h*)(qbase + hi * 16);
  a0.p.hi = *(const v8h*)(qbase + 8 + hi * 16);
  a1.p.lo = *(const v8h*)(qbase + 32 + hi * 16);
  a1.p.hi = *(const v8h*)(qbase + 40 + hi * 16);

  // pad columns 272..287 (never produced by the 17 tiles)
  #pragma unroll
  for (int r = 0; r < 8; ++r) P[r + 8 * hi][272 + m] = NEGI;

  // ---- S = q @ ckf^T over 17 key tiles ----
  for (int jt = 0; jt < 17; ++jt) {
    const _Float16* cb = ckf + ((size_t)b * NCPAD + jt * 16 + m) * KPAD;
    if (jt + 1 < 17)
      __builtin_prefetch(ckf + ((size_t)b * NCPAD + (jt + 1) * 16 + m) * KPAD, 0, 1);
    v16h b0 = *(const v16h*)(cb + hi * 16);
    v16h b1 = *(const v16h*)(cb + 32 + hi * 16);
    v8f c = {};
    c = WMMA_F16(a0.v, b0, c);
    c = WMMA_F16(a1.v, b1, c);
    int j = jt * 16 + m;
    #pragma unroll
    for (int r = 0; r < 8; ++r) {
      int row = qt * 16 + r + 8 * hi;        // global sequence position
      float s = c[r] * SCALEF;
      bool keep = (j == 0) || (j <= NBLK && row >= j * CBS);  // causal block mask
      P[r + 8 * hi][j] = keep ? s : NEGI;
    }
  }
  __syncthreads();

  // ---- per-row softmax + argmax compressed block (=> top-k via repeat ties) ----
  {
    int row  = lane >> 1;
    int half = lane & 1;
    int j0 = half * (NCPAD / 2);
    float vmax = NEGI, bv = NEGI;
    int bj = 0;
    for (int j = j0; j < j0 + NCPAD / 2; ++j) {
      float s = P[row][j];
      vmax = fmaxf(vmax, s);
      if (j >= 1 && j <= NBLK && s > bv) { bv = s; bj = j - 1; }
    }
    float ov = __shfl_xor(vmax, 1);
    vmax = fmaxf(vmax, ov);
    float obv = __shfl_xor(bv, 1);
    int   obj = __shfl_xor(bj, 1);
    if (obv > bv || (obv == bv && obj < bj)) { bv = obv; bj = obj; }
    float sum = 0.f;
    for (int j = j0; j < j0 + NCPAD / 2; ++j) {
      float pz = __expf(P[row][j] - vmax);
      P[row][j] = pz;
      sum += pz;
    }
    sum += __shfl_xor(sum, 1);
    if (half == 0) {
      denomS[row] = sum;
      float sv = __expf(bv - vmax) / sum;
      size_t gi = (size_t)b * NSEQ + qt * 16 + row;
      selJb[gi] = bj;
      selOk[gi] = (sv > 1e-10f) ? 1 : 0;
    }
  }
  __syncthreads();

  // ---- c_out = P @ cvf   (K = 288 in 9 steps, N = 48 in 3 tiles) ----
  v8f acc0 = {}, acc1 = {}, acc2 = {};
  for (int ts = 0; ts < NCPAD / 32; ++ts) {
    V16U a;
    #pragma unroll
    for (int u = 0; u < 8; ++u) a.h[u]     = (_Float16)P[m][ts * 32 + hi * 16 + u];
    #pragma unroll
    for (int u = 0; u < 8; ++u) a.h[8 + u] = (_Float16)P[m][ts * 32 + 8 + hi * 16 + u];
    v16h bb0 = *(const v16h*)(cvfT + ((size_t)b * DH +  0 + m) * NCPAD + ts * 32 + hi * 16);
    v16h bb1 = *(const v16h*)(cvfT + ((size_t)b * DH + 16 + m) * NCPAD + ts * 32 + hi * 16);
    v16h bb2 = *(const v16h*)(cvfT + ((size_t)b * DH + 32 + m) * NCPAD + ts * 32 + hi * 16);
    acc0 = WMMA_F16(a.v, bb0, acc0);
    acc1 = WMMA_F16(a.v, bb1, acc1);
    acc2 = WMMA_F16(a.v, bb2, acc2);
  }
  #pragma unroll
  for (int r = 0; r < 8; ++r) {
    int rloc = r + 8 * hi;
    float inv = 1.0f / denomS[rloc];
    size_t ob = ((size_t)b * NSEQ + qt * 16 + rloc) * DH;
    c_out[ob +  0 + m] = acc0[r] * inv;
    c_out[ob + 16 + m] = acc1[r] * inv;
    c_out[ob + 32 + m] = acc2[r] * inv;
  }
}

// ---------------------------------------------------------------------------
// Kernel D: fine attention + sliding window + gating + block partial reduce
// ---------------------------------------------------------------------------
__global__ __launch_bounds__(256) void k_fine(
    const float* __restrict__ qr, const float* __restrict__ kr,
    const float* __restrict__ vf, const float* __restrict__ c_out,
    const float* __restrict__ gg, const int* __restrict__ selJb,
    const int* __restrict__ selOk, float* __restrict__ partial)
{
  const int BPB = NSEQ / 256;                // 12 blocks per batch
  int b   = blockIdx.x / BPB;
  int blk = blockIdx.x % BPB;
  int tid = threadIdx.x;
  int i   = blk * 256 + tid;
  size_t gi = (size_t)b * NSEQ + i;

  float qv[DH];
  #pragma unroll
  for (int d = 0; d < DH; ++d) qv[d] = qr[gi * DH + d];

  float g0 = gg[gi * 3 + 0], g1 = gg[gi * 3 + 1], g2 = gg[gi * 3 + 2];
  float o[DH];
  #pragma unroll
  for (int d = 0; d < DH; ++d) o[d] = g0 * c_out[gi * DH + d];

  // --- fine attention: small blocks {6*jb, 6*jb+1, own} ---
  int jb = selJb[gi];
  int ok = selOk[gi];
  int sblk[3] = {6 * jb, 6 * jb + 1, i / 2};
  int bok[3]  = {ok, ok, 1};
  int pk[6]; float sim[6];
  float fmaxv = NEGI;
  #pragma unroll
  for (int t = 0; t < 3; ++t) {
    #pragma unroll
    for (int s = 0; s < 2; ++s) {
      int p = sblk[t] * 2 + s;
      pk[t * 2 + s] = p;
      float sv = NEGI;
      if (bok[t] && p <= i) {
        float a = 0.f;
        for (int d = 0; d < DH; ++d) a += qv[d] * kr[((size_t)b * NSEQ + p) * DH + d];
        sv = a * SCALEF;
      }
      sim[t * 2 + s] = sv;
      fmaxv = fmaxf(fmaxv, sv);
    }
  }
  float fden = 0.f;
  #pragma unroll
  for (int e = 0; e < 6; ++e) { sim[e] = __expf(sim[e] - fmaxv); fden += sim[e]; }
  float finv = g1 / fden;
  #pragma unroll
  for (int e = 0; e < 6; ++e) {
    float w = sim[e] * finv;
    const float* vp = vf + ((size_t)b * NSEQ + pk[e]) * DH;
    for (int d = 0; d < DH; ++d) o[d] += w * vp[d];
  }

  // --- sliding window (7 positions) ---
  float ss[WIN + 1];
  float smax = NEGI;
  #pragma unroll
  for (int w = 0; w <= WIN; ++w) {
    int p = i - w;
    float sv = NEGI;
    if (p >= 0) {
      float a = 0.f;
      for (int d = 0; d < DH; ++d) a += qv[d] * kr[((size_t)b * NSEQ + p) * DH + d];
      sv = a * SCALEF;
    }
    ss[w] = sv; smax = fmaxf(smax, sv);
  }
  float sden = 0.f;
  #pragma unroll
  for (int w = 0; w <= WIN; ++w) { ss[w] = __expf(ss[w] - smax); sden += ss[w]; }
  float sinv = g2 / sden;
  #pragma unroll
  for (int w = 0; w <= WIN; ++w) {
    int p = i - w;
    if (p >= 0) {
      float wgt = ss[w] * sinv;
      const float* vp = vf + ((size_t)b * NSEQ + p) * DH;
      for (int d = 0; d < DH; ++d) o[d] += wgt * vp[d];
    }
  }

  // --- deterministic block reduction of o over the 256 tokens ---
  __shared__ float red[256][DH];
  #pragma unroll
  for (int d = 0; d < DH; ++d) red[tid][d] = o[d];
  __syncthreads();
  for (int s = 128; s > 0; s >>= 1) {
    if (tid < s)
      for (int d = 0; d < DH; ++d) red[tid][d] += red[tid + s][d];
    __syncthreads();
  }
  if (tid < DH) partial[((size_t)b * BPB + blk) * DH + tid] = red[0][tid];
}

// ---------------------------------------------------------------------------
// Kernel F: mean-pool through Wo, MLP head (exact GELU), final output
// ---------------------------------------------------------------------------
__global__ __launch_bounds__(64) void k_head(
    const float* __restrict__ partial, const float* __restrict__ Wo,
    const float* __restrict__ Wh1, const float* __restrict__ bh1,
    const float* __restrict__ Wh2, const float* __restrict__ bh2,
    float* __restrict__ out)
{
  const int BPB = NSEQ / 256;
  int b = blockIdx.x;
  int t = threadIdx.x;
  __shared__ float obar[DH];
  __shared__ float pooled[DIMM];
  __shared__ float hdn[32];
  if (t < DH) {
    float a = 0.f;
    for (int c = 0; c < BPB; ++c) a += partial[((size_t)b * BPB + c) * DH + t];
    obar[t] = a * (1.0f / NSEQ);
  }
  __syncthreads();
  {
    float a = 0.f;
    for (int d = 0; d < DH; ++d) a += obar[d] * Wo[d * DIMM + t];
    pooled[t] = a;
  }
  __syncthreads();
  if (t < 32) {
    float a = bh1[t];
    for (int e = 0; e < DIMM; ++e) a += pooled[e] * Wh1[e * 32 + t];
    hdn[t] = 0.5f * a * (1.0f + erff(a * 0.70710678118654752f));   // exact GELU
  }
  __syncthreads();
  if (t == 0) {
    float a = bh2[0];
    for (int j = 0; j < 32; ++j) a += hdn[j] * Wh2[j];
    out[b] = a;
  }
}

// ---------------------------------------------------------------------------
extern "C" void kernel_launch(void* const* d_in, const int* in_sizes, int n_in,
                              void* d_out, int out_size, void* d_ws, size_t ws_size,
                              hipStream_t stream)
{
  (void)in_sizes; (void)n_in; (void)out_size; (void)ws_size;
  const float* X     = (const float*)d_in[0];
  const float* w_emb = (const float*)d_in[1];
  const float* b_emb = (const float*)d_in[2];
  const float* gamma = (const float*)d_in[3];
  const float* Wqkv  = (const float*)d_in[4];
  const float* k_pos = (const float*)d_in[5];
  const float* v_pos = (const float*)d_in[6];
  const float* mem_k = (const float*)d_in[7];
  const float* mem_v = (const float*)d_in[8];
  const float* Wkc   = (const float*)d_in[9];
  const float* bkc   = (const float*)d_in[10];
  const float* Wvc   = (const float*)d_in[11];
  const float* bvc   = (const float*)d_in[12];
  const float* Wg    = (const float*)d_in[13];
  const float* bg    = (const float*)d_in[14];
  const float* Wo    = (const float*)d_in[15];
  const float* Wh1   = (const float*)d_in[16];
  const float* bh1   = (const float*)d_in[17];
  const float* Wh2   = (const float*)d_in[18];
  const float* bh2   = (const float*)d_in[19];
  float* out = (float*)d_out;

  char* p = (char*)d_ws;
  auto alloc = [&](size_t bytes) -> char* {
    char* r = p;
    p += (bytes + 255) & ~(size_t)255;
    return r;
  };
  const size_t BN = (size_t)BBATCH * NSEQ;
  _Float16* xnh   = (_Float16*)alloc(BN * DIMM * sizeof(_Float16));
  _Float16* qh    = (_Float16*)alloc(BN * KPAD * sizeof(_Float16));
  float* qr       = (float*)alloc(BN * DH * sizeof(float));
  float* kr       = (float*)alloc(BN * DH * sizeof(float));
  float* vf       = (float*)alloc(BN * DH * sizeof(float));
  float* gg       = (float*)alloc(BN * 3 * sizeof(float));
  _Float16* kbh   = (_Float16*)alloc((size_t)BBATCH * NBLK * CBD * sizeof(_Float16));
  _Float16* vbh   = (_Float16*)alloc((size_t)BBATCH * NBLK * CBD * sizeof(_Float16));
  _Float16* ckf   = (_Float16*)alloc((size_t)BBATCH * NCPAD * KPAD * sizeof(_Float16));
  _Float16* cvfT  = (_Float16*)alloc((size_t)BBATCH * DH * NCPAD * sizeof(_Float16));
  float* c_out    = (float*)alloc(BN * DH * sizeof(float));
  int* selJb      = (int*)alloc(BN * sizeof(int));
  int* selOk      = (int*)alloc(BN * sizeof(int));
  float* partial  = (float*)alloc((size_t)BBATCH * (NSEQ / 256) * DH * sizeof(float));
  _Float16* WqkvT = (_Float16*)alloc((size_t)NQKV * DIMM * sizeof(_Float16));
  _Float16* WkcT  = (_Float16*)alloc((size_t)DH * CBD * sizeof(_Float16));
  _Float16* WvcT  = (_Float16*)alloc((size_t)DH * CBD * sizeof(_Float16));

  const int prepElems = NQKV * DIMM + 2 * DH * CBD;
  k_prepw<<<(prepElems + 255) / 256, 256, 0, stream>>>(Wqkv, Wkc, Wvc,
                                                       WqkvT, WkcT, WvcT);
  k_embed<<<(int)(BN / 256), 256, 0, stream>>>(X, w_emb, b_emb, gamma,
                                               xnh, Wg, bg, gg);
  k_qkv<<<BBATCH * (NSEQ / 16), 32, 0, stream>>>(xnh, WqkvT, k_pos, v_pos,
                                                 qh, qr, kr, vf, kbh, vbh);
  k_initc<<<BBATCH, NCPAD, 0, stream>>>(mem_k, mem_v, ckf, cvfT);
  k_compress<<<BBATCH * (NBLK / 16), 32, 0, stream>>>(kbh, vbh, WkcT, WvcT,
                                                      bkc, bvc, ckf, cvfT);
  k_cattn<<<BBATCH * (NSEQ / 16), 32, 0, stream>>>(qh, ckf, cvfT, c_out, selJb, selOk);
  k_fine<<<BBATCH * (NSEQ / 256), 256, 0, stream>>>(qr, kr, vf, c_out, gg,
                                                    selJb, selOk, partial);
  k_head<<<BBATCH, 64, 0, stream>>>(partial, Wo, Wh1, bh1, Wh2, bh2, out);
}
